// ShiftWindowMSA_78898549228102
// MI455X (gfx1250) — matrix operand order repaired and checked by reading, not compiled, not run on hardware
//
#include <hip/hip_runtime.h>

// ---------------- problem constants (match reference) ----------------
#define BQ      32
#define HH      56
#define CCH     384
#define NHEADS  12
#define DHEAD   32        // head dim
#define WS7     7
#define SS3     3
#define NWIN    49        // tokens per window
#define NWINP   64        // padded tokens
#define BW      2048      // BQ * 64 windows
#define MROWS   (BW * NWIN)   // 100352  (divisible by 128)
#define KDIM    384
#define NOUT    1152      // 3*C

typedef _Float16 half_t;
typedef _Float16 v8h  __attribute__((ext_vector_type(8)));
typedef _Float16 v16h __attribute__((ext_vector_type(16)));
typedef float    v8f  __attribute__((ext_vector_type(8)));

// Gather: window-partition of roll(x, -3, -3).  Row m of the flattened
// (Bw*N) matrix comes from x[b, (wy*7+i+3)%56, (wx*7+j+3)%56, :].
__device__ __forceinline__ int src_row_index(int m) {
    int bw  = m / NWIN;
    int n   = m - bw * NWIN;
    int b   = bw >> 6;
    int win = bw & 63;
    int wy = win >> 3, wx = win & 7;
    int i = n / 7, j = n - i * 7;
    int hp = wy * 7 + i + SS3; if (hp >= HH) hp -= HH;
    int wp = wx * 7 + j + SS3; if (wp >= HH) wp -= HH;
    return b * (HH * HH) + hp * HH + wp;
}

__device__ __forceinline__ void cvt_store16(half_t* dst, const float4 f[4]) {
    #pragma unroll
    for (int q4 = 0; q4 < 4; ++q4) {
        dst[q4 * 4 + 0] = (half_t)f[q4].x;
        dst[q4 * 4 + 1] = (half_t)f[q4].y;
        dst[q4 * 4 + 2] = (half_t)f[q4].z;
        dst[q4 * 4 + 3] = (half_t)f[q4].w;
    }
}

// =====================================================================
// Kernel 1: fused gather + QKV GEMM (f32 -> f16 WMMA -> f16 qkv ws)
//   out = gather(x) @ qkv_w^T + qkv_b
//   q,k stored (Bw, h, N, d); v stored transposed+padded (Bw, h, d, 64)
// Block: 256 threads (8 waves). Tile: M=128, N=64, K step 32.
// K-loop is register double-buffered: global prefetch of step i+1 issues
// before the WMMAs of step i, hiding HBM/L2 latency under matrix work.
// =====================================================================
__global__ __launch_bounds__(256)
void qkv_gemm_kernel(const float* __restrict__ x,
                     const float* __restrict__ w,
                     const float* __restrict__ bias,
                     half_t* __restrict__ qws,
                     half_t* __restrict__ kws,
                     half_t* __restrict__ vws) {
    __shared__ half_t As[128 * 32];   // row-major M x K  (8 KB)
    __shared__ half_t Bs[64 * 32];    // Bs[col][k] = w[col][k] (4 KB)

    const int tid  = threadIdx.x;
    const int wave = tid >> 5;
    const int lane = tid & 31;
    const int mbase = blockIdx.y * 128;
    const int nbase = blockIdx.x * 64;

    // A staging: 2 threads per row, 16 floats each
    const int arow  = tid >> 1;
    const int apart = tid & 1;
    const float* xsrc = x + (size_t)src_row_index(mbase + arow) * CCH + apart * 16;
    // B staging: 4 threads per output column, 8 floats each
    const int bj    = tid >> 2;
    const int bpart = tid & 3;
    const float* wsrc = w + (size_t)(nbase + bj) * KDIM + bpart * 8;

    v8f acc[4] = {};
    float4 aR[4], bR[2];

    {   // prologue prefetch (k-step 0)
        const float4* pa = (const float4*)xsrc;
        #pragma unroll
        for (int q4 = 0; q4 < 4; ++q4) aR[q4] = pa[q4];
        const float4* pb = (const float4*)wsrc;
        bR[0] = pb[0]; bR[1] = pb[1];
    }

    const int ar = wave * 16 + (lane & 15);
    const int k0 = (lane >> 4) * 16;    // A half-lane K chunk base
    const int kb = (lane >> 4) * 16;    // B half-lane contiguous K base

    for (int kk = 0; kk < KDIM; kk += 32) {
        __syncthreads();                 // prev-step frag reads complete
        // stage current step from registers (f32 -> f16, packed cvt)
        cvt_store16(&As[arow * 32 + apart * 16], aR);
        {
            half_t* db = &Bs[bj * 32 + bpart * 8];
            db[0] = (half_t)bR[0].x; db[1] = (half_t)bR[0].y;
            db[2] = (half_t)bR[0].z; db[3] = (half_t)bR[0].w;
            db[4] = (half_t)bR[1].x; db[5] = (half_t)bR[1].y;
            db[6] = (half_t)bR[1].z; db[7] = (half_t)bR[1].w;
        }
        // prefetch next step while this step computes
        if (kk + 32 < KDIM) {
            const float4* pa = (const float4*)(xsrc + kk + 32);
            #pragma unroll
            for (int q4 = 0; q4 < 4; ++q4) aR[q4] = pa[q4];
            const float4* pb = (const float4*)(wsrc + kk + 32);
            bR[0] = pb[0]; bR[1] = pb[1];
        }
        __syncthreads();

        // A fragment (16x32 f16): half-lane contiguous k-chunk {k0..k0+15}
        v8h alo = *(const v8h*)&As[ar * 32 + k0];
        v8h ahi = *(const v8h*)&As[ar * 32 + k0 + 8];
        v16h afrag = __builtin_shufflevector(alo, ahi,
            0, 1, 2, 3, 4, 5, 6, 7, 8, 9, 10, 11, 12, 13, 14, 15);

        // preload all B fragments, then issue WMMAs back-to-back
        v16h bfrag[4];
        #pragma unroll
        for (int ct = 0; ct < 4; ++ct)
            bfrag[ct] = *(const v16h*)&Bs[(ct * 16 + (lane & 15)) * 32 + kb];
        #pragma unroll
        for (int ct = 0; ct < 4; ++ct)
            acc[ct] = __builtin_amdgcn_wmma_f32_16x16x32_f16(
                false, afrag, false, bfrag[ct], (short)0, acc[ct], false, false);
    }

    // Epilogue: bias + scatter to q/k/(v transposed, padded) f16 workspace.
    #pragma unroll
    for (int ct = 0; ct < 4; ++ct) {
        const int j = nbase + ct * 16 + (lane & 15);
        const int which = j / CCH;          // 0=q 1=k 2=v
        const int c = j - which * CCH;
        const int head = c >> 5, dc = c & 31;
        const float bv = bias[j];
        #pragma unroll
        for (int r = 0; r < 8; ++r) {
            const int m = mbase + wave * 16 + ((lane >> 4) << 3) + r;
            const int bw = m / NWIN;
            const int n  = m - bw * NWIN;
            const half_t hv = (half_t)(acc[ct][r] + bv);
            const size_t wh = (size_t)(bw * NHEADS + head);
            if (which == 0)       qws[(wh * NWIN + n) * DHEAD + dc] = hv;
            else if (which == 1)  kws[(wh * NWIN + n) * DHEAD + dc] = hv;
            else                  vws[(wh * DHEAD + dc) * NWINP + n] = hv;
        }
    }
}

// =====================================================================
// Kernel 2: per-(window, head) attention + LePE + fused window-reverse.
// Block: 128 threads (4 waves); wave i owns S/O row tile i.
// q/k tiles are moved into LDS with gfx1250 async-to-LDS DMA
// (ASYNCcnt-tracked), bypassing the VGPR round trip.
// =====================================================================
__global__ __launch_bounds__(128)
void win_attn_kernel(const half_t* __restrict__ qws,
                     const half_t* __restrict__ kws,
                     const half_t* __restrict__ vws,
                     const float* __restrict__ lepe_w,
                     const float* __restrict__ lepe_b,
                     float* __restrict__ out) {
    __shared__ half_t sQ[NWINP * DHEAD];     // 64x32 f16
    __shared__ half_t sK[NWINP * DHEAD];     // 64x32 f16
    __shared__ half_t sVT[DHEAD * NWINP];    // 32x64 f16 (v transposed)
    __shared__ float  sS[NWINP * NWINP];     // 64x64 f32 scores
    __shared__ half_t sP[NWINP * NWINP];     // 64x64 f16 probs
    __shared__ float  sO[NWINP * DHEAD];     // 64x32 f32
    __shared__ float  sLw[DHEAD * 9];
    __shared__ float  sLb[DHEAD];
    __shared__ int    sCnt[NWINP];

    const int tid  = threadIdx.x;
    const int wave = tid >> 5;
    const int lane = tid & 31;
    const int bw   = blockIdx.x;
    const int head = blockIdx.y;
    const int bno = bw >> 6, win = bw & 63;
    const int wy = win >> 3, wx = win & 7;

    // ---- phase 1: async q/k DMA + zero pads + LePE params + mask ids ----
    {   // q/k: linear 16B chunks, (49*32*2)/16 = 196 transfers each
        const size_t qoff = ((size_t)bw * NHEADS + head) * (NWIN * DHEAD);
        const unsigned long long qbase = (unsigned long long)(uintptr_t)(qws + qoff);
        const unsigned long long kbase = (unsigned long long)(uintptr_t)(kws + qoff);
        const unsigned ldq0 = (unsigned)(uintptr_t)sQ;
        const unsigned ldk0 = (unsigned)(uintptr_t)sK;
        for (int i2 = tid; i2 < (NWIN * DHEAD) / 8; i2 += 128) {
            const unsigned go = (unsigned)i2 * 16u;
            asm volatile("global_load_async_to_lds_b128 %0, %1, %2"
                         :: "v"(ldq0 + go), "v"(go), "s"(qbase) : "memory");
            asm volatile("global_load_async_to_lds_b128 %0, %1, %2"
                         :: "v"(ldk0 + go), "v"(go), "s"(kbase) : "memory");
        }
    }
    for (int i2 = tid; i2 < NWINP * NWINP; i2 += 128) sP[i2] = (half_t)0.f;
    for (int i2 = NWIN * DHEAD + tid; i2 < NWINP * DHEAD; i2 += 128) {
        sQ[i2] = (half_t)0.f;
        sK[i2] = (half_t)0.f;
    }
    {   // v^T: padded (32,64) layout in ws -> vectorized copy, zero pad cols
        const v8h* vs = (const v8h*)(vws +
            ((size_t)bw * NHEADS + head) * (DHEAD * NWINP));
        v8h* vd = (v8h*)sVT;
        for (int i2 = tid; i2 < (DHEAD * NWINP) / 8; i2 += 128) {
            v8h h = vs[i2];
            const int c0 = (i2 & 7) * 8;     // token col of element 0
            #pragma unroll
            for (int e = 0; e < 8; ++e)
                if (c0 + e >= NWIN) h[e] = (half_t)0.f;
            vd[i2] = h;
        }
    }
    for (int i2 = tid; i2 < DHEAD * 9; i2 += 128)
        sLw[i2] = lepe_w[head * (DHEAD * 9) + i2];
    if (tid < DHEAD) sLb[tid] = lepe_b[head * DHEAD + tid];
    if (tid < NWINP) {
        int cnt = 0;
        if (tid < NWIN) {
            const int i = tid / 7, j = tid - i * 7;
            const int hp = wy * 7 + i, wp = wx * 7 + j;
            const int rh = hp < 49 ? 0 : (hp < 53 ? 1 : 2);
            const int rw = wp < 49 ? 0 : (wp < 53 ? 1 : 2);
            cnt = rh * 3 + rw;
        }
        sCnt[tid] = cnt;
    }
    asm volatile("s_wait_asynccnt 0x0" ::: "memory");
    __syncthreads();

    // ---- phase 2: S = q @ k^T  (wave -> row tile, 4 col tiles) ----
    {
        const int ar = wave * 16 + (lane & 15);
        const int k0 = (lane >> 4) * 16;
        v8h alo = *(const v8h*)&sQ[ar * DHEAD + k0];
        v8h ahi = *(const v8h*)&sQ[ar * DHEAD + k0 + 8];
        v16h a = __builtin_shufflevector(alo, ahi,
            0, 1, 2, 3, 4, 5, 6, 7, 8, 9, 10, 11, 12, 13, 14, 15);
        const int kbq = (lane >> 4) * 16;
        v16h b[4];
        #pragma unroll
        for (int ct = 0; ct < 4; ++ct)
            b[ct] = *(const v16h*)&sK[(ct * 16 + (lane & 15)) * DHEAD + kbq];
        v8f c[4] = {};
        #pragma unroll
        for (int ct = 0; ct < 4; ++ct)
            c[ct] = __builtin_amdgcn_wmma_f32_16x16x32_f16(
                false, a, false, b[ct], (short)0, c[ct], false, false);
        #pragma unroll
        for (int ct = 0; ct < 4; ++ct) {
            const int col = ct * 16 + (lane & 15);
            #pragma unroll
            for (int r = 0; r < 8; ++r)
                sS[(wave * 16 + ((lane >> 4) << 3) + r) * NWINP + col] = c[ct][r];
        }
    }
    __syncthreads();

    // ---- phase 3: masked softmax per row (scale folded in here) ----
    if (tid < NWIN) {
        const float scale = 0.17677669529663687f;   // 32^-0.5
        float* row = &sS[tid * NWINP];
        const int cm = sCnt[tid];
        float mx = -1e30f;
        for (int m2 = 0; m2 < NWIN; ++m2) {
            float v = row[m2] * scale + (sCnt[m2] == cm ? 0.f : -100.f);
            row[m2] = v;
            mx = fmaxf(mx, v);
        }
        float s = 0.f;
        for (int m2 = 0; m2 < NWIN; ++m2) {
            float e = __expf(row[m2] - mx);
            row[m2] = e;
            s += e;
        }
        const float inv = 1.f / s;
        for (int m2 = 0; m2 < NWIN; ++m2)
            sP[tid * NWINP + m2] = (half_t)(row[m2] * inv);
    }
    __syncthreads();

    // ---- phase 4: O = P @ V  (K=64 in 2 steps, 2 d-col tiles) ----
    {
        v8f o[2] = {};
        #pragma unroll
        for (int ks2 = 0; ks2 < 2; ++ks2) {
            const int ar = wave * 16 + (lane & 15);
            const int k0 = ks2 * 32 + (lane >> 4) * 16;
            v8h alo = *(const v8h*)&sP[ar * NWINP + k0];
            v8h ahi = *(const v8h*)&sP[ar * NWINP + k0 + 8];
            v16h a = __builtin_shufflevector(alo, ahi,
                0, 1, 2, 3, 4, 5, 6, 7, 8, 9, 10, 11, 12, 13, 14, 15);
            const int kbp = ks2 * 32 + (lane >> 4) * 16;
            v16h b[2];
            #pragma unroll
            for (int ct = 0; ct < 2; ++ct)
                b[ct] = *(const v16h*)&sVT[(ct * 16 + (lane & 15)) * NWINP + kbp];
            #pragma unroll
            for (int ct = 0; ct < 2; ++ct)
                o[ct] = __builtin_amdgcn_wmma_f32_16x16x32_f16(
                    false, a, false, b[ct], (short)0, o[ct], false, false);
        }
        #pragma unroll
        for (int ct = 0; ct < 2; ++ct) {
            const int col = ct * 16 + (lane & 15);
            #pragma unroll
            for (int r = 0; r < 8; ++r)
                sO[(wave * 16 + ((lane >> 4) << 3) + r) * DHEAD + col] = o[ct][r];
        }
    }
    __syncthreads();

    // ---- phase 5: LePE depthwise 3x3 + fused window-reverse scatter ----
    for (int e = tid; e < NWIN * DHEAD; e += 128) {
        const int n = e >> 5, c = e & 31;
        const int i = n / 7, j = n - i * 7;
        float acc = sLb[c];
        #pragma unroll
        for (int ky = 0; ky < 3; ++ky)
            #pragma unroll
            for (int kx = 0; kx < 3; ++kx) {
                const int ii = i + ky - 1, jj = j + kx - 1;
                if (ii >= 0 && ii < 7 && jj >= 0 && jj < 7)
                    acc += sLw[c * 9 + ky * 3 + kx]
                         * (float)sVT[c * NWINP + ii * 7 + jj];
            }
        const float val = sO[n * DHEAD + c] + acc;
        int hp = wy * 7 + i + SS3; if (hp >= HH) hp -= HH;
        int wp = wx * 7 + j + SS3; if (wp >= HH) wp -= HH;
        out[((size_t)bno * (HH * HH) + hp * HH + wp) * CCH + head * DHEAD + c] = val;
    }
}

// =====================================================================
// Launch: inputs = {query, qkv_w, qkv_b, lepe_w, lepe_b, H, W}
// Workspace: q,k (Bw,h,49,32) + v (Bw,h,32,64) f16 buffers (~255 MB).
// =====================================================================
extern "C" void kernel_launch(void* const* d_in, const int* in_sizes, int n_in,
                              void* d_out, int out_size, void* d_ws, size_t ws_size,
                              hipStream_t stream) {
    const float* query  = (const float*)d_in[0];
    const float* qkv_w  = (const float*)d_in[1];
    const float* qkv_b  = (const float*)d_in[2];
    const float* lepe_w = (const float*)d_in[3];
    const float* lepe_b = (const float*)d_in[4];
    float* out = (float*)d_out;

    const size_t qn = (size_t)BW * NHEADS * NWIN * DHEAD;    // 38,535,168 halfs
    const size_t vn = (size_t)BW * NHEADS * DHEAD * NWINP;   // 50,331,648 halfs
    half_t* qws = (half_t*)d_ws;
    half_t* kws = qws + qn;
    half_t* vws = kws + qn;
    (void)vn;

    dim3 g1(NOUT / 64, MROWS / 128);   // (18, 784): N-tiles fast -> A-row L2 reuse
    qkv_gemm_kernel<<<g1, 256, 0, stream>>>(query, qkv_w, qkv_b, qws, kws, vws);

    dim3 g2(BW, NHEADS);               // (2048, 12)
    win_attn_kernel<<<g2, 128, 0, stream>>>(qws, kws, vws, lepe_w, lepe_b, out);
}
